// MultiHeadAttention_13383118094900
// MI455X (gfx1250) — compile-verified, matrix operand order checked by reading, not run on hardware
//
#include <hip/hip_runtime.h>
#include <hip/hip_bf16.h>
#include <math.h>
#include <stdint.h>

#define BB  4
#define SS  2048
#define DD  1024
#define HH  16
#define DHH 64

typedef __attribute__((ext_vector_type(16))) _Float16 v16h;
typedef __attribute__((ext_vector_type(8)))  _Float16 v8h;
typedef __attribute__((ext_vector_type(8)))  float    v8f;
typedef __attribute__((ext_vector_type(4)))  unsigned int u32x4;
typedef __attribute__((ext_vector_type(8)))  int          i32x8;
typedef __attribute__((ext_vector_type(4)))  int          i32x4;

__device__ __forceinline__ v8f wmma_f16(v16h a, v16h b, v8f c) {
  // D = A(16x32 f16) * B(32x16 f16) + C(16x16 f32)
  return __builtin_amdgcn_wmma_f32_16x16x32_f16(false, a, false, b, (short)0, c, false, false);
}

#if __has_builtin(__builtin_amdgcn_tensor_load_to_lds) && __has_builtin(__builtin_amdgcn_s_wait_tensorcnt)
#define USE_TDM 1
#else
#define USE_TDM 0
#endif

#if USE_TDM
// Build a 2-D TDM descriptor (D#) and issue TENSOR_LOAD_TO_LDS.
// dims/tiles/strides in data_size (2-byte) units. Layout per ISA cdna5 ch.8:
//  g0: [1:0]=count=1 | lds_addr | global_addr[56:0] | type=2
//  g1: data_size=1(2B), tensor_dim0/1, tile_dim0/1/2, tensor_dim0_stride
// This toolchain exposes the 6-arg builtin: (g0, g1, g2, g3, g4, cpol).
__device__ __forceinline__ void tdm_load_2d(unsigned lds_off_bytes, const void* gptr,
                                            unsigned dim0, unsigned dim1,
                                            unsigned tile0, unsigned tile1,
                                            unsigned stride0) {
  unsigned long long ga = (unsigned long long)(uintptr_t)gptr;
  u32x4 g0;
  g0[0] = 1u;                                            // count=1, user descriptor
  g0[1] = lds_off_bytes;                                 // lds_addr
  g0[2] = (unsigned)(ga & 0xffffffffu);                  // global_addr[31:0]
  g0[3] = (unsigned)((ga >> 32) & 0x01ffffffu) | (2u << 30);  // addr[56:32] | type=2
  i32x8 g1;
  g1[0] = (int)(1u << 16);                               // wg_mask=0, data_size=1 (2B)
  g1[1] = (int)((dim0 & 0xffffu) << 16);                 // tensor_dim0[15:0]
  g1[2] = (int)(((dim0 >> 16) & 0xffffu) | ((dim1 & 0xffffu) << 16));
  g1[3] = (int)(((dim1 >> 16) & 0xffffu) | ((tile0 & 0xffffu) << 16));
  g1[4] = (int)(tile1 & 0xffffu);                        // tile_dim1, tile_dim2=0
  g1[5] = (int)stride0;                                  // tensor_dim0_stride[31:0]
  g1[6] = 0;
  g1[7] = 0;
  i32x4 z4 = {0, 0, 0, 0};
  i32x8 z8 = {0, 0, 0, 0, 0, 0, 0, 0};
  __builtin_amdgcn_tensor_load_to_lds(g0, g1, z4, z4, z8, 0);
}
#endif

// ---------------- conversion kernels ----------------
__global__ void cvt_f32_to_f16(const float* __restrict__ in, _Float16* __restrict__ out, int n) {
  int i = blockIdx.x * blockDim.x + threadIdx.x;
  if (i < n) out[i] = (_Float16)in[i];
}

// Wt[n][k] = W[k][n]  (so WMMA B-fragment columns are contiguous)
__global__ void cvt_w_transpose(const float* __restrict__ Wq, const float* __restrict__ Wk,
                                const float* __restrict__ Wv, const float* __restrict__ Wo,
                                _Float16* __restrict__ WtQ, _Float16* __restrict__ WtK,
                                _Float16* __restrict__ WtV, _Float16* __restrict__ WtO) {
  int i = blockIdx.x * blockDim.x + threadIdx.x;   // over DD*DD
  const float* W  = (blockIdx.y == 0) ? Wq : (blockIdx.y == 1) ? Wk : (blockIdx.y == 2) ? Wv : Wo;
  _Float16*    Wt = (blockIdx.y == 0) ? WtQ : (blockIdx.y == 1) ? WtK : (blockIdx.y == 2) ? WtV : WtO;
  int nn = i >> 10, kk = i & (DD - 1);
  Wt[i] = (_Float16)W[kk * DD + nn];
}

// ---------------- fused QKV projection GEMM (4M x 2N register blocking) ----
__global__ __launch_bounds__(256) void gemm_qkv(
    const _Float16* __restrict__ xh,
    const _Float16* __restrict__ WtQ, const _Float16* __restrict__ WtK, const _Float16* __restrict__ WtV,
    const float* __restrict__ bq, const float* __restrict__ bk, const float* __restrict__ bv,
    _Float16* __restrict__ Qh, _Float16* __restrict__ Kh, _Float16* __restrict__ Vt) {
  const int mode = blockIdx.y;
  const int wave = threadIdx.x >> 5;
  const int lane = threadIdx.x & 31;
  const int n = lane & 15, hi = lane >> 4;
  const int tilesN = DD / 32;
  const int job = blockIdx.x * 8 + wave;
  const int m0 = (job / tilesN) * 64;
  const int n0 = (job % tilesN) * 32;

  const _Float16* wt   = (mode == 0) ? WtQ : (mode == 1) ? WtK : WtV;
  const float*    bias = (mode == 0) ? bq  : (mode == 1) ? bk  : bv;

  const _Float16* arow[4];
  const _Float16* brow[2];
#pragma unroll
  for (int mt = 0; mt < 4; ++mt) arow[mt] = xh + (size_t)(m0 + mt * 16 + n) * DD;
#pragma unroll
  for (int nt = 0; nt < 2; ++nt) brow[nt] = wt + (size_t)(n0 + nt * 16 + n) * DD;

  v8f acc[4][2] = {};
  for (int k = 0; k < DD; k += 32) {
    v16h a[4], bf[2];
#pragma unroll
    for (int mt = 0; mt < 4; ++mt) {
      ((v8h*)&a[mt])[0] = *(const v8h*)(arow[mt] + k + hi * 8);
      ((v8h*)&a[mt])[1] = *(const v8h*)(arow[mt] + k + 16 + hi * 8);
    }
#pragma unroll
    for (int nt = 0; nt < 2; ++nt) {
      ((v8h*)&bf[nt])[0] = *(const v8h*)(brow[nt] + k + hi * 8);
      ((v8h*)&bf[nt])[1] = *(const v8h*)(brow[nt] + k + 16 + hi * 8);
    }
    __builtin_prefetch(arow[0] + k + 512, 0, 3);   // global_prefetch_b8
    __builtin_prefetch(brow[0] + k + 512, 0, 3);
#pragma unroll
    for (int mt = 0; mt < 4; ++mt)
#pragma unroll
      for (int nt = 0; nt < 2; ++nt)
        acc[mt][nt] = wmma_f16(a[mt], bf[nt], acc[mt][nt]);
  }

#pragma unroll
  for (int nt = 0; nt < 2; ++nt) {
    const int col = n0 + nt * 16 + n;
    const int h = col >> 6, dh = col & 63;
    const float bval = bias[col];
#pragma unroll
    for (int mt = 0; mt < 4; ++mt)
#pragma unroll
      for (int r = 0; r < 8; ++r) {
        int row = m0 + mt * 16 + r + 8 * hi;
        int b = row >> 11, s = row & (SS - 1);
        float v = acc[mt][nt][r] + bval;
        if (mode == 0) {
          v *= 0.125f;                              // 1/sqrt(64) folded into Q
          Qh[((size_t)(b * HH + h) * SS + s) * DHH + dh] = (_Float16)v;
        } else if (mode == 1) {
          Kh[((size_t)(b * HH + h) * SS + s) * DHH + dh] = (_Float16)v;
        } else {
          Vt[((size_t)(b * HH + h) * DHH + dh) * SS + s] = (_Float16)v;
        }
      }
  }
}

// ---------------- flash attention with TDM-staged K/V -----------------------
// Block = 8 waves * 32 queries = 256 queries, all sharing (b,h). K(32x64) and
// V^T(64x32) tiles for each 32-key block are DMA'd once per block into
// double-buffered LDS by the Tensor Data Mover (wave 0 issues, TENSORcnt
// ordered), instead of 8 waves redundantly streaming them from L2.
__global__ __launch_bounds__(256) void flash_attn(
    const _Float16* __restrict__ Qh, const _Float16* __restrict__ Kh,
    const _Float16* __restrict__ Vt, _Float16* __restrict__ ctx) {
  extern __shared__ _Float16 smem[];                // 2 * (2048 K + 2048 V) halves
  const int tid  = threadIdx.x;
  const int wave = tid >> 5;
  const int lane = tid & 31;
  const int n = lane & 15, hi = lane >> 4;
  const int q0_blk = blockIdx.x * 256;
  const int q_base = q0_blk + wave * 32;            // this wave's first query
  const int h = blockIdx.y, b = blockIdx.z;
  const size_t bh = (size_t)(b * HH + h);
  const _Float16* Qb = Qh + bh * SS * DHH;
  const _Float16* Kb = Kh + bh * SS * DHH;
  const _Float16* Vb = Vt + bh * DHH * SS;

  // Q as B-fragments (col = q), 2 q-tiles x 2 DH chunks of K=32
  v16h qf[2][2];
#pragma unroll
  for (int qt = 0; qt < 2; ++qt) {
    const _Float16* p0 = Qb + (size_t)(q_base + qt * 16 + n) * DHH + hi * 8;
    ((v8h*)&qf[qt][0])[0] = *(const v8h*)(p0);
    ((v8h*)&qf[qt][0])[1] = *(const v8h*)(p0 + 16);
    ((v8h*)&qf[qt][1])[0] = *(const v8h*)(p0 + 32);
    ((v8h*)&qf[qt][1])[1] = *(const v8h*)(p0 + 48);
  }

  v8f acc[2][4] = {};
  float m[2]    = {-3.0e38f, -3.0e38f};
  float lsum[2] = {0.f, 0.f};

  const int nIter = (q0_blk + 256) / 32;            // key blocks needed by top wave

#if USE_TDM
  if (wave == 0) {                                  // prologue: stage block 0 into buf 0
    tdm_load_2d(0,    Kb, DHH, SS, DHH, 32, DHH);
    tdm_load_2d(4096, Vb, SS, DHH, 32, DHH, SS);
  }
#else
  {                                                 // cooperative fallback staging
    int r = tid >> 3, o = (tid & 7) * 8;
    *(v8h*)(smem + r * DHH + o) = *(const v8h*)(Kb + (size_t)r * DHH + o);
    int r2 = tid >> 2, o2 = (tid & 3) * 8;
    *(v8h*)(smem + 2048 + r2 * 32 + o2) = *(const v8h*)(Vb + (size_t)r2 * SS + o2);
  }
#endif

  for (int i = 0; i < nIter; ++i) {
    const int c = i * 32;
    const unsigned buf = (unsigned)(i & 1);
#if USE_TDM
    if (wave == 0) {
      if (i + 1 < nIter) {                          // stage next block into other buffer
        const unsigned nb = (unsigned)((i + 1) & 1) * 8192u;
        tdm_load_2d(nb,        Kb + (size_t)(c + 32) * DHH, DHH, SS - (c + 32), DHH, 32, DHH);
        tdm_load_2d(nb + 4096, Vb + (c + 32),               SS - (c + 32), DHH, 32, DHH, SS);
        __builtin_amdgcn_s_wait_tensorcnt(2);       // current buffer's pair done
      } else {
        __builtin_amdgcn_s_wait_tensorcnt(0);
      }
    }
#else
    if (i + 1 < nIter) {
      int nb = ((i + 1) & 1) * 4096;
      int r = tid >> 3, o = (tid & 7) * 8;
      *(v8h*)(smem + nb + r * DHH + o) = *(const v8h*)(Kb + (size_t)(c + 32 + r) * DHH + o);
      int r2 = tid >> 2, o2 = (tid & 3) * 8;
      *(v8h*)(smem + nb + 2048 + r2 * 32 + o2) = *(const v8h*)(Vb + (size_t)r2 * SS + c + 32 + o2);
    }
#endif
    __syncthreads();                                // staged buffer visible to all waves

    if (c <= q_base + 31) {                         // wave-uniform causal activity test
      const _Float16* Ksm = smem + buf * 4096;      // [key_local][dh]  32x64
      const _Float16* Vsm = Ksm + 2048;             // [dh][key_local]  64x32

      // K tiles as A-fragments from LDS (rows = keys), shared by both q-tiles
      v16h k00, k01, k10, k11;
      {
        const _Float16* kp0 = Ksm + n * DHH + hi * 8;
        ((v8h*)&k00)[0] = *(const v8h*)(kp0);
        ((v8h*)&k00)[1] = *(const v8h*)(kp0 + 16);
        ((v8h*)&k01)[0] = *(const v8h*)(kp0 + 32);
        ((v8h*)&k01)[1] = *(const v8h*)(kp0 + 48);
        const _Float16* kp1 = kp0 + 16 * DHH;
        ((v8h*)&k10)[0] = *(const v8h*)(kp1);
        ((v8h*)&k10)[1] = *(const v8h*)(kp1 + 16);
        ((v8h*)&k11)[0] = *(const v8h*)(kp1 + 32);
        ((v8h*)&k11)[1] = *(const v8h*)(kp1 + 48);
      }

      v16h pf[2];
#pragma unroll
      for (int qt = 0; qt < 2; ++qt) {
        const int q = q_base + qt * 16 + n;
        v8f st0 = {}, st1 = {};
        st0 = wmma_f16(k00, qf[qt][0], st0);
        st0 = wmma_f16(k01, qf[qt][1], st0);
        st1 = wmma_f16(k10, qf[qt][0], st1);
        st1 = wmma_f16(k11, qf[qt][1], st1);

        float s0[8], s1[8];
        float mt = -3.0e38f;
#pragma unroll
        for (int r = 0; r < 8; ++r) {
          int key0 = c + r + 8 * hi;
          float a0 = (key0 <= q) ? st0[r] : -3.0e38f;
          float a1 = (key0 + 16 <= q) ? st1[r] : -3.0e38f;
          s0[r] = a0; s1[r] = a1;
          mt = fmaxf(mt, fmaxf(a0, a1));
        }
        mt = fmaxf(mt, __shfl_xor(mt, 16, 32));
        float mnew = fmaxf(m[qt], mt);
        float corr = __expf(m[qt] - mnew);

        v16h p = {};
        float psum = 0.f;
#pragma unroll
        for (int r = 0; r < 8; ++r) {
          float p0 = __expf(s0[r] - mnew);
          float p1 = __expf(s1[r] - mnew);
          psum += p0 + p1;
          p[r]     = (_Float16)p0;
          p[r + 8] = (_Float16)p1;
        }
        psum += __shfl_xor(psum, 16, 32);
        lsum[qt] = lsum[qt] * corr + psum;
#pragma unroll
        for (int t = 0; t < 4; ++t)
#pragma unroll
          for (int r = 0; r < 8; ++r) acc[qt][t][r] *= corr;
        pf[qt] = p;
        m[qt] = mnew;
      }

      // C^T += V^T_tile (A-frag from LDS) * P^T (B-frag in registers)
#pragma unroll
      for (int t = 0; t < 4; ++t) {
        v16h vf;
        const _Float16* vp = Vsm + (t * 16 + n) * 32 + hi * 8;
        ((v8h*)&vf)[0] = *(const v8h*)(vp);
        ((v8h*)&vf)[1] = *(const v8h*)(vp + 16);
        acc[0][t] = wmma_f16(vf, pf[0], acc[0][t]);
        acc[1][t] = wmma_f16(vf, pf[1], acc[1][t]);
      }
    }
    __syncthreads();                                // all reads done before next overwrite
  }

#pragma unroll
  for (int qt = 0; qt < 2; ++qt) {
    const int q = q_base + qt * 16 + n;
    const float inv = 1.f / lsum[qt];
#pragma unroll
    for (int t = 0; t < 4; ++t)
#pragma unroll
      for (int r = 0; r < 8; ++r) {
        int dh = t * 16 + r + 8 * hi;
        ctx[((size_t)(b * SS + q) * HH + h) * DHH + dh] = (_Float16)(acc[qt][t][r] * inv);
      }
  }
}

// ---------------- output projection GEMM (4M x 2N blocked, f32 out) -------
__global__ __launch_bounds__(256) void gemm_out(
    const _Float16* __restrict__ ctx, const _Float16* __restrict__ WtO,
    const float* __restrict__ bo, float* __restrict__ out) {
  const int wave = threadIdx.x >> 5;
  const int lane = threadIdx.x & 31;
  const int n = lane & 15, hi = lane >> 4;
  const int tilesN = DD / 32;
  const int job = blockIdx.x * 8 + wave;
  const int m0 = (job / tilesN) * 64;
  const int n0 = (job % tilesN) * 32;

  const _Float16* arow[4];
  const _Float16* brow[2];
#pragma unroll
  for (int mt = 0; mt < 4; ++mt) arow[mt] = ctx + (size_t)(m0 + mt * 16 + n) * DD;
#pragma unroll
  for (int nt = 0; nt < 2; ++nt) brow[nt] = WtO + (size_t)(n0 + nt * 16 + n) * DD;

  v8f acc[4][2] = {};
  for (int k = 0; k < DD; k += 32) {
    v16h a[4], bf[2];
#pragma unroll
    for (int mt = 0; mt < 4; ++mt) {
      ((v8h*)&a[mt])[0] = *(const v8h*)(arow[mt] + k + hi * 8);
      ((v8h*)&a[mt])[1] = *(const v8h*)(arow[mt] + k + 16 + hi * 8);
    }
#pragma unroll
    for (int nt = 0; nt < 2; ++nt) {
      ((v8h*)&bf[nt])[0] = *(const v8h*)(brow[nt] + k + hi * 8);
      ((v8h*)&bf[nt])[1] = *(const v8h*)(brow[nt] + k + 16 + hi * 8);
    }
    __builtin_prefetch(arow[0] + k + 512, 0, 3);
    __builtin_prefetch(brow[0] + k + 512, 0, 3);
#pragma unroll
    for (int mt = 0; mt < 4; ++mt)
#pragma unroll
      for (int nt = 0; nt < 2; ++nt)
        acc[mt][nt] = wmma_f16(a[mt], bf[nt], acc[mt][nt]);
  }

#pragma unroll
  for (int nt = 0; nt < 2; ++nt) {
    const int col = n0 + nt * 16 + n;
    const float bval = bo[col];
#pragma unroll
    for (int mt = 0; mt < 4; ++mt)
#pragma unroll
      for (int r = 0; r < 8; ++r) {
        int row = m0 + mt * 16 + r + 8 * hi;
        out[(size_t)row * DD + col] = acc[mt][nt][r] + bval;
      }
  }
}

extern "C" void kernel_launch(void* const* d_in, const int* in_sizes, int n_in,
                              void* d_out, int out_size, void* d_ws, size_t ws_size,
                              hipStream_t stream) {
  (void)in_sizes; (void)n_in; (void)out_size; (void)ws_size;
  const float* x  = (const float*)d_in[0];
  const float* Wq = (const float*)d_in[1];
  const float* bq = (const float*)d_in[2];
  const float* Wk = (const float*)d_in[3];
  const float* bk = (const float*)d_in[4];
  const float* Wv = (const float*)d_in[5];
  const float* bv = (const float*)d_in[6];
  const float* Wo = (const float*)d_in[7];
  const float* bo = (const float*)d_in[8];
  float* out = (float*)d_out;

  const size_t BS = (size_t)BB * SS;
  const size_t nX = BS * DD;
  const size_t nW = (size_t)DD * DD;

  _Float16* xh  = (_Float16*)d_ws;
  _Float16* WtQ = xh  + nX;
  _Float16* WtK = WtQ + nW;
  _Float16* WtV = WtK + nW;
  _Float16* WtO = WtV + nW;
  _Float16* Qh  = WtO + nW;
  _Float16* Kh  = Qh  + nX;
  _Float16* Vt  = Kh  + nX;
  _Float16* ctx = Vt  + nX;

  cvt_f32_to_f16<<<dim3((unsigned)(nX / 256)), 256, 0, stream>>>(x, xh, (int)nX);
  cvt_w_transpose<<<dim3((unsigned)(nW / 256), 4), 256, 0, stream>>>(Wq, Wk, Wv, Wo, WtQ, WtK, WtV, WtO);

  gemm_qkv<<<dim3(512, 3), 256, 0, stream>>>(xh, WtQ, WtK, WtV, bq, bk, bv, Qh, Kh, Vt);

  // 8 waves/block, 256 queries/block; 16 KiB dynamic LDS (double-buffered K/V tiles)
  flash_attn<<<dim3(SS / 256, HH, BB), 256, 16384, stream>>>(Qh, Kh, Vt, ctx);

  gemm_out<<<dim3(512), 256, 0, stream>>>(ctx, WtO, bo, out);
}
